// ParaAttention_14740327760298
// MI455X (gfx1250) — matrix operand order repaired and checked by reading, not compiled, not run on hardware
//
#include <hip/hip_runtime.h>

typedef __attribute__((ext_vector_type(16))) _Float16 v16h;
typedef __attribute__((ext_vector_type(8)))  _Float16 v8h;
typedef __attribute__((ext_vector_type(8)))  float    v8f;
typedef __attribute__((ext_vector_type(4)))  float    v4f;
typedef __attribute__((ext_vector_type(4)))  unsigned int v4u;
typedef __attribute__((ext_vector_type(8)))  int      v8i;
typedef __attribute__((ext_vector_type(4)))  int      v4i;
typedef _Float16 h16;

#define DIMC   512
#define NTOK   4096
#define BBATCH 8
#define QSCALE 0.125f   // 64^-0.5

#if __has_builtin(__builtin_amdgcn_tensor_load_to_lds) && __has_builtin(__builtin_amdgcn_s_wait_tensorcnt)
#define USE_TDM 1
#else
#define USE_TDM 0
#endif

// ---------------------------------------------------------------------------
// WMMA helpers (CDNA5 wave32, 16x16x32 f16 -> f32)
// ---------------------------------------------------------------------------
__device__ __forceinline__ v8f wmma_f16(v16h a, v16h b, v8f c) {
  return __builtin_amdgcn_wmma_f32_16x16x32_f16(
      false, a, false, b, (short)0, c, false, false);
}

// A fragment: logical A[m][k], m in [0,16), k in [0,32), row-major, stride lda.
// Lane L holds row M=L&15; lane half selects K chunk (L>>4)*16 -> 16 contiguous halves.
// Both 8-half chunks are contiguous -> 2x ds_load_b128.
__device__ __forceinline__ v16h frag_a(const h16* A, int lda) {
  const int lane = threadIdx.x & 31;
  const h16* row = A + (lane & 15) * lda + ((lane >> 4) * 16);
  v16h f;
#pragma unroll
  for (int e = 0; e < 8; ++e) f[e] = row[e];
#pragma unroll
  for (int e = 0; e < 8; ++e) f[8 + e] = row[8 + e];
  return f;
}

// B fragment from transposed storage: logical B[k][n] held as Bt[n][k].
// Lane L holds column N=L&15, K chunk (L>>4)*16 -> 16 contiguous halves.
__device__ __forceinline__ v16h frag_bt(const h16* Bt, int ldb) {
  const int lane = threadIdx.x & 31;
  const h16* row = Bt + (lane & 15) * ldb + ((lane >> 4) * 16);
  v16h f;
#pragma unroll
  for (int e = 0; e < 16; ++e) f[e] = row[e];
  return f;
}

// ---------------------------------------------------------------------------
// Weight pack: conv w (O,512,3,3) f32 -> wpk[tap][o][c] f16  (K-contiguous)
// ---------------------------------------------------------------------------
__global__ void pack_w_kernel(const float* __restrict__ w, h16* __restrict__ out, int O) {
  int idx = blockIdx.x * 256 + threadIdx.x;
  if (idx >= 9 * 512 * O) return;
  int c   = idx & 511;
  int rem = idx >> 9;
  int o   = rem % O;
  int tap = rem / O;
  out[idx] = (h16)w[((size_t)o * 512 + c) * 9 + tap];
}

// ---------------------------------------------------------------------------
// LayerNorm over C=512 per token; write f16 in [b][j][i][c] layout.
// ---------------------------------------------------------------------------
__global__ void __launch_bounds__(256) ln_kernel(const float* __restrict__ x,
                                                 const float* __restrict__ g,
                                                 const float* __restrict__ bt,
                                                 h16* __restrict__ out) {
  __shared__ float red[256];
  const int tid = threadIdx.x;
  const int blk = blockIdx.x;          // b*4096 + t, t = i*64 + j
  const int b = blk >> 12;
  const int t = blk & 4095;
  const int i = t >> 6;
  const int j = t & 63;
  const float* xp = x + (size_t)blk * DIMC;
  float v0 = xp[tid], v1 = xp[tid + 256];
  red[tid] = v0 + v1; __syncthreads();
#pragma unroll
  for (int o = 128; o > 0; o >>= 1) { if (tid < o) red[tid] += red[tid + o]; __syncthreads(); }
  float mu = red[0] * (1.0f / DIMC);
  __syncthreads();
  float d0 = v0 - mu, d1 = v1 - mu;
  red[tid] = d0 * d0 + d1 * d1; __syncthreads();
#pragma unroll
  for (int o = 128; o > 0; o >>= 1) { if (tid < o) red[tid] += red[tid + o]; __syncthreads(); }
  float rstd = rsqrtf(red[0] * (1.0f / DIMC) + 1e-5f);
  h16* op = out + ((((size_t)b * 64 + j) * 64 + i) << 9);
  op[tid]       = (h16)(d0 * rstd * g[tid]       + bt[tid]);
  op[tid + 256] = (h16)(d1 * rstd * g[tid + 256] + bt[tid + 256]);
}

// ---------------------------------------------------------------------------
// Spatial-mixing MLP:  Y = Wb(64x128) @ gelu( Wa(128x64) @ X(64x512) )
// X and H staged transposed in LDS so B fragments are vector ds loads.
// in : [b][outer][axis][c]   out: [b][axis'][outer][c]  (transposed write)
// ---------------------------------------------------------------------------
#define LDX 72     // padded ld for sXT  [cc=64][i=64]
#define LDH 136    // padded ld for sHT  [cc=64][o=128]
__global__ void __launch_bounds__(256) mix_mlp_kernel(const h16* __restrict__ in,
                                                      h16* __restrict__ out,
                                                      const float* __restrict__ wa,
                                                      const float* __restrict__ wb) {
  __shared__ h16 sWa[128 * 64];
  __shared__ h16 sWb[64 * 128];
  __shared__ h16 sXT[64 * LDX];
  __shared__ h16 sHT[64 * LDH];
  const int tid    = threadIdx.x;
  const int bx     = blockIdx.x;
  const int cchunk = bx & 7;
  const int row    = bx >> 3;          // b*64 + outer
  const int outer  = row & 63;
  const int b      = row >> 6;
  const int cbase  = cchunk << 6;

#pragma unroll
  for (int it = 0; it < 8; ++it) {     // 2048 float4 per weight matrix
    int idx = tid + it * 256;
    v4f a4 = ((const v4f*)wa)[idx];
    v4f b4 = ((const v4f*)wb)[idx];
#pragma unroll
    for (int e = 0; e < 4; ++e) { sWa[idx * 4 + e] = (h16)a4[e]; sWb[idx * 4 + e] = (h16)b4[e]; }
  }
  const h16* ip = in + ((size_t)row << 15) + cbase;   // row * 64 * 512
#pragma unroll
  for (int it = 0; it < 2; ++it) {     // 512 v8h chunks, transposed scatter
    int idx = tid + it * 256;
    int i = idx >> 3, kc = idx & 7;
    v8h hv = *(const v8h*)(ip + ((size_t)i << 9) + kc * 8);
#pragma unroll
    for (int e = 0; e < 8; ++e) sXT[(kc * 8 + e) * LDX + i] = hv[e];
  }
  __syncthreads();

  const int w    = tid >> 5;
  const int lane = tid & 31;
  const int n0   = lane & 15;
  const int mb   = (lane >> 4) << 3;

  // GEMM1: H(128x64) = Wa(128x64) @ X(64x64chunk), fused exact GELU
  {
    v16h a0 = frag_a(&sWa[(w << 4) * 64], 64);
    v16h a1 = frag_a(&sWa[(w << 4) * 64 + 32], 64);
#pragma unroll
    for (int nt = 0; nt < 4; ++nt) {
      v8f acc = {};
      acc = wmma_f16(a0, frag_bt(&sXT[(nt << 4) * LDX], LDX), acc);
      acc = wmma_f16(a1, frag_bt(&sXT[(nt << 4) * LDX + 32], LDX), acc);
#pragma unroll
      for (int r = 0; r < 8; ++r) {
        float xv = acc[r];
        float gv = 0.5f * xv * (1.0f + erff(xv * 0.70710678f));
        sHT[((nt << 4) + n0) * LDH + (w << 4) + mb + r] = (h16)gv;  // transposed
      }
    }
  }
  __syncthreads();

  // GEMM2: Y(64x64chunk) = Wb(64x128) @ H(128x64chunk); write transposed
  {
    const int mt  = w >> 1;
    const int ntb = (w & 1) << 1;
    v16h a[4];
#pragma unroll
    for (int ks = 0; ks < 4; ++ks) a[ks] = frag_a(&sWb[(mt << 4) * 128 + (ks << 5)], 128);
#pragma unroll
    for (int tn = 0; tn < 2; ++tn) {
      int nt = ntb + tn;
      v8f acc = {};
#pragma unroll
      for (int ks = 0; ks < 4; ++ks)
        acc = wmma_f16(a[ks], frag_bt(&sHT[(nt << 4) * LDH + (ks << 5)], LDH), acc);
#pragma unroll
      for (int r = 0; r < 8; ++r) {
        int aout = (mt << 4) + mb + r;
        int ncol = (nt << 4) + n0;
        out[((((size_t)b * 64 + aout) * 64 + outer) << 9) + cbase + ncol] = (h16)acc[r];
      }
    }
  }
}

// ---------------------------------------------------------------------------
// 3x3 conv (pad 1) as implicit GEMM, f16 WMMA, f32 accumulate.
// in  : f16 [b][s0][s1][512]      wpk : f16 [tap][o][c]   bias: f32 [O]
// Dynamic LDS: sIn [3][66][32] @ 0, sW [tap][n][k] @ 12672.
// Weight tile per K-chunk fetched by the Tensor Data Mover when available:
// 3-D descriptor x=32 ch (contig), y=64 outch (stride 512), z=9 taps
// (stride O*512); TDM (z,y,x) fill order == sW layout.
// ---------------------------------------------------------------------------
#define SIN_HALVES (3 * 66 * 32)              // 6336
#define SW_OFF_B   (SIN_HALVES * 2)           // 12672 bytes
#define CONV_LDS_B (SW_OFF_B + 9 * 64 * 32 * 2)  // 49536 bytes
__global__ void __launch_bounds__(256) conv3x3_kernel(const h16* __restrict__ in,
                                                      const h16* __restrict__ wpk,
                                                      const float* __restrict__ bias,
                                                      h16* __restrict__ outh,
                                                      float* __restrict__ outf,
                                                      int O, int scale_q) {
  extern __shared__ char smem[];
  h16* sIn = (h16*)smem;                 // [r][p][k]
  h16* sW  = (h16*)(smem + SW_OFF_B);    // [tap][n][k]
  const int tid = threadIdx.x;
  const int row = blockIdx.x;            // b*64 + s0
  const int s0  = row & 63;
  const int b   = row >> 6;
  const int ob  = blockIdx.y << 6;
  const int w   = tid >> 5;
  const int mt  = w & 3;
  const int ng  = w >> 2;
  v8f acc[2]; acc[0] = {}; acc[1] = {};

  for (int cc = 0; cc < 16; ++cc) {
    if (cc + 1 < 16)
      __builtin_prefetch(&in[((((size_t)b * 64 + s0) * 64) << 9) + ((cc + 1) << 5)], 0, 1);
    __syncthreads();
#if USE_TDM
    if (tid < 32) {
      unsigned long long ga =
          (unsigned long long)(const void*)(wpk + ((size_t)ob << 9) + ((size_t)cc << 5));
      v4u g0;
      g0[0] = 1u;                                           // count=1
      g0[1] = (unsigned)SW_OFF_B;                           // lds_addr (bytes)
      g0[2] = (unsigned)(ga & 0xffffffffu);                 // global_addr lo
      g0[3] = (unsigned)((ga >> 32) & 0x01ffffffu) | 0x80000000u;  // hi | type=2
      unsigned s1 = (unsigned)O << 9;                       // dim1 stride (elems)
      v8i g1;
      g1[0] = 1 << 16;                                      // data_size = 2B
      g1[1] = 512 << 16;                                    // tensor_dim0 = 512
      g1[2] = 64 << 16;                                     // tensor_dim1 = 64
      g1[3] = 32 << 16;                                     // tile_dim0 = 32
      g1[4] = 64 | (9 << 16);                               // tile_dim1=64, tile_dim2=9
      g1[5] = 512;                                          // tensor_dim0_stride
      g1[6] = (int)((s1 & 0xffffu) << 16);                  // dim1_stride lo16
      g1[7] = (int)(s1 >> 16);                              // dim1_stride hi
      v4i g2; g2[0] = 9; g2[1] = 0; g2[2] = 0; g2[3] = 0;   // tensor_dim2 = 9
      v4i g3; g3[0] = 0; g3[1] = 0; g3[2] = 0; g3[3] = 0;
      v8i g4 = {};                                          // unused extra group
      __builtin_amdgcn_tensor_load_to_lds(g0, g1, g2, g3, g4, 0);
    }
#else
    for (int idx = tid; idx < 9 * 64 * 4; idx += 256) {     // 2304 v8h chunks
      int tap = idx >> 8, rem = idx & 255;
      int n = rem >> 2, kc = rem & 3;
      v8h hv = *(const v8h*)(wpk + (((size_t)tap * O + ob + n) << 9) + (cc << 5) + kc * 8);
      *(v8h*)&sW[(tap << 11) + (n << 5) + kc * 8] = hv;
    }
#endif
    for (int idx = tid; idx < 3 * 66 * 4; idx += 256) {     // 792 v8h chunks
      int kc = idx & 3;
      int p  = (idx >> 2) % 66;
      int r  = (idx >> 2) / 66;
      int s0g = s0 + r - 1;
      int s1g = p - 1;
      v8h hv = {};
      if ((unsigned)s0g < 64u && (unsigned)s1g < 64u)
        hv = *(const v8h*)(in + ((((size_t)b * 64 + s0g) * 64 + s1g) << 9) + (cc << 5) + kc * 8);
      *(v8h*)&sIn[((r * 66 + p) << 5) + kc * 8] = hv;
    }
#if USE_TDM
    if (tid < 32) __builtin_amdgcn_s_wait_tensorcnt(0);
#endif
    __syncthreads();
#pragma unroll
    for (int tap = 0; tap < 9; ++tap) {
      const int dy = tap / 3, dx = tap % 3;
      v16h a = frag_a(&sIn[((dy * 66) + (mt << 4) + dx) << 5], 32);
      acc[0] = wmma_f16(a, frag_bt(&sW[(tap << 11) + ((((ng << 1) + 0) << 4) << 5)], 32), acc[0]);
      acc[1] = wmma_f16(a, frag_bt(&sW[(tap << 11) + ((((ng << 1) + 1) << 4) << 5)], 32), acc[1]);
    }
  }

  const int lane = tid & 31;
  const int n0   = lane & 15;
  const int mb   = (lane >> 4) << 3;
#pragma unroll
  for (int tn = 0; tn < 2; ++tn) {
    int o = ob + (((ng << 1) + tn) << 4) + n0;
    float bv = bias[o];
#pragma unroll
    for (int r = 0; r < 8; ++r) {
      int s1 = (mt << 4) + mb + r;
      float val = acc[tn][r] + bv;
      if (scale_q && o < 512) val *= QSCALE;       // fold q scaling into conv
      size_t tIdx = (size_t)b * NTOK + s0 * 64 + s1;
      if (outf) outf[tIdx * 512 + o] = val;
      else      outh[tIdx * (size_t)O + o] = (h16)val;
    }
  }
}

// ---------------------------------------------------------------------------
// Windowed attention: one block per (b, head, partition). Q/K/V are 64x64.
// S = Q K^T, fp32 softmax, O = P V.  V staged transposed for vector B frags.
// ---------------------------------------------------------------------------
#define LDV 72
__global__ void __launch_bounds__(256) attn_kernel(const h16* __restrict__ qkv,
                                                   h16* __restrict__ out,
                                                   int transpose_out) {
  __shared__ h16  sQ[64 * 64];
  __shared__ h16  sK[64 * 64];
  __shared__ h16  sVT[64 * LDV];
  __shared__ float sS[64 * 64];
  __shared__ h16  sP[64 * 64];
  const int tid = threadIdx.x;
  const int blk = blockIdx.x;          // (b*8 + h)*64 + p
  const int p = blk & 63;
  const int h = (blk >> 6) & 7;
  const int b = blk >> 9;
  const size_t base = ((size_t)b * NTOK + p * 64) * 1536 + h * 64;
#pragma unroll
  for (int it = 0; it < 2; ++it) {     // 512 v8h chunks
    int idx = tid + it * 256;
    int l = idx >> 3, kc = idx & 7;
    size_t g = base + (size_t)l * 1536 + kc * 8;
    *(v8h*)&sQ[(l << 6) + kc * 8] = *(const v8h*)(qkv + g);
    *(v8h*)&sK[(l << 6) + kc * 8] = *(const v8h*)(qkv + g + 512);
    v8h hv = *(const v8h*)(qkv + g + 1024);
#pragma unroll
    for (int e = 0; e < 8; ++e) sVT[(kc * 8 + e) * LDV + l] = hv[e];
  }
  __syncthreads();

  const int w    = tid >> 5;
  const int mt   = w >> 1;
  const int ntb  = (w & 1) << 1;
  const int lane = tid & 31;
  const int n0   = lane & 15;
  const int mb   = (lane >> 4) << 3;

  // S = Q @ K^T   (B operand = K, naturally transposed: Bt[n=l'][k=d])
  {
    v16h a0 = frag_a(&sQ[(mt << 4) * 64], 64);
    v16h a1 = frag_a(&sQ[(mt << 4) * 64 + 32], 64);
#pragma unroll
    for (int tn = 0; tn < 2; ++tn) {
      int nt = ntb + tn;
      v8f acc = {};
      acc = wmma_f16(a0, frag_bt(&sK[(nt << 4) * 64], 64), acc);
      acc = wmma_f16(a1, frag_bt(&sK[(nt << 4) * 64 + 32], 64), acc);
#pragma unroll
      for (int r = 0; r < 8; ++r)
        sS[((mt << 4) + mb + r) * 64 + (nt << 4) + n0] = acc[r];
    }
  }
  __syncthreads();

  if (tid < 64) {                      // fp32 softmax, one row per thread
    const float* rowp = &sS[tid * 64];
    float m = -1e30f;
#pragma unroll
    for (int c = 0; c < 64; ++c) m = fmaxf(m, rowp[c]);
    float s = 0.0f;
#pragma unroll
    for (int c = 0; c < 64; ++c) s += expf(rowp[c] - m);
    float inv = 1.0f / s;
#pragma unroll
    for (int c = 0; c < 64; ++c) sP[tid * 64 + c] = (h16)(expf(rowp[c] - m) * inv);
  }
  __syncthreads();

  // O = P @ V  (B operand staged transposed: Bt[n=d][k=l'])
  {
    v16h a0 = frag_a(&sP[(mt << 4) * 64], 64);
    v16h a1 = frag_a(&sP[(mt << 4) * 64 + 32], 64);
#pragma unroll
    for (int tn = 0; tn < 2; ++tn) {
      int nt = ntb + tn;
      v8f acc = {};
      acc = wmma_f16(a0, frag_bt(&sVT[(nt << 4) * LDV], LDV), acc);
      acc = wmma_f16(a1, frag_bt(&sVT[(nt << 4) * LDV + 32], LDV), acc);
      int cch = h * 64 + (nt << 4) + n0;
#pragma unroll
      for (int r = 0; r < 8; ++r) {
        int l = (mt << 4) + mb + r;
        size_t oidx = transpose_out
            ? ((((size_t)b * 64 + l) * 64 + p) << 9) + cch
            : ((((size_t)b * 64 + p) * 64 + l) << 9) + cch;
        out[oidx] = (h16)acc[r];
      }
    }
  }
}

// ---------------------------------------------------------------------------
extern "C" void kernel_launch(void* const* d_in, const int* in_sizes, int n_in,
                              void* d_out, int out_size, void* d_ws, size_t ws_size,
                              hipStream_t stream) {
  (void)in_sizes; (void)n_in; (void)out_size; (void)ws_size;
  const float* x      = (const float*)d_in[0];
  const float* qkv_w  = (const float*)d_in[1];
  const float* qkv_b  = (const float*)d_in[2];
  const float* proj_w = (const float*)d_in[3];
  const float* proj_b = (const float*)d_in[4];
  const float* ln_g   = (const float*)d_in[5];
  const float* ln_b   = (const float*)d_in[6];
  const float* m1a = (const float*)d_in[7];
  const float* m1b = (const float*)d_in[8];
  const float* m2a = (const float*)d_in[9];
  const float* m2b = (const float*)d_in[10];
  const float* m3a = (const float*)d_in[11];
  const float* m3b = (const float*)d_in[12];
  const float* m4a = (const float*)d_in[13];
  const float* m4b = (const float*)d_in[14];

  char* ws = (char*)d_ws;
  size_t off = 0;
  auto alloc = [&](size_t bytes) -> char* {
    char* pp = ws + off; off += (bytes + 255) & ~(size_t)255; return pp;
  };
  h16* actA = (h16*)alloc((size_t)BBATCH * NTOK * DIMC * 2);           // 33.5 MB
  h16* actB = (h16*)alloc((size_t)BBATCH * NTOK * DIMC * 2);           // 33.5 MB
  h16* qkvb = (h16*)alloc((size_t)BBATCH * NTOK * 1536 * 2);           // 100.7 MB
  h16* wq   = (h16*)alloc((size_t)9 * 512 * 1536 * 2);                 // 14.2 MB
  h16* wp   = (h16*)alloc((size_t)9 * 512 * 512 * 2);                  // 4.7 MB

  pack_w_kernel<<<(9 * 512 * 1536 + 255) / 256, 256, 0, stream>>>(qkv_w, wq, 1536);
  pack_w_kernel<<<(9 * 512 * 512 + 255) / 256, 256, 0, stream>>>(proj_w, wp, 512);

  // token mix: LN -> 4 alternating spatial MLPs (ping-pong, transposed writes)
  ln_kernel<<<BBATCH * NTOK, 256, 0, stream>>>(x, ln_g, ln_b, actA);
  mix_mlp_kernel<<<4096, 256, 0, stream>>>(actA, actB, m1a, m1b);
  mix_mlp_kernel<<<4096, 256, 0, stream>>>(actB, actA, m2a, m2b);
  mix_mlp_kernel<<<4096, 256, 0, stream>>>(actA, actB, m3a, m3b);
  mix_mlp_kernel<<<4096, 256, 0, stream>>>(actB, actA, m4a, m4b);

  // attention pass 1 (QKV conv -> windowed attention, transposed writeback)
  conv3x3_kernel<<<dim3(512, 24), 256, CONV_LDS_B, stream>>>(actA, wq, qkv_b, qkvb, nullptr, 1536, 1);
  attn_kernel<<<4096, 256, 0, stream>>>(qkvb, actB, 1);

  // attention pass 2 (straight writeback feeds proj conv)
  conv3x3_kernel<<<dim3(512, 24), 256, CONV_LDS_B, stream>>>(actB, wq, qkv_b, qkvb, nullptr, 1536, 1);
  attn_kernel<<<4096, 256, 0, stream>>>(qkvb, actA, 0);

  // proj conv, fp32 output straight to d_out (B, N, C)
  conv3x3_kernel<<<dim3(512, 8), 256, CONV_LDS_B, stream>>>(actA, wp, proj_b, nullptr,
                                                            (float*)d_out, 512, 0);
}
